// Sinkhornknopp_56573309223312
// MI455X (gfx1250) — matrix operand order repaired and checked by reading, not compiled, not run on hardware
//
#include <hip/hip_runtime.h>

// Sinkhorn-Knopp, n=m=8192, d=256, eps=0.1.
// The reference's loop uses the CONSTANT b in `u = a/(K@b+eps)`, so (u,v) hit a
// fixed point after one iteration -> compute u,v once.
// K (fp32, 256MB) lives in d_out as scratch and is rescaled in place.
// Fast path (ws_size >= ~8.6MB): pre-convert inputs to f16, 32x64 tiles/wave.
// Fallback (ws_size >= 160KB): inline fp32->f16 conversion, 32x32 tiles/wave.

#define NN 8192
#define DD 256

typedef __attribute__((ext_vector_type(16))) _Float16 v16h;
typedef __attribute__((ext_vector_type(8)))  _Float16 v8h;
typedef __attribute__((ext_vector_type(4)))  _Float16 v4h;
typedef __attribute__((ext_vector_type(8)))  float    v8f;

// ---------------------------------------------------------------- row norms
__global__ void __launch_bounds__(256) rownorm2_kernel(
    const float* __restrict__ X, const float* __restrict__ Y,
    float* __restrict__ x2, float* __restrict__ y2) {
  const float* src = blockIdx.y ? Y : X;
  float* dst       = blockIdx.y ? y2 : x2;
  int wave = threadIdx.x >> 5;
  int lane = threadIdx.x & 31;
  int row  = blockIdx.x * 8 + wave;
  const float* p = src + (size_t)row * DD;
  float s = 0.f;
#pragma unroll
  for (int t = 0; t < DD / 32; ++t) {
    float v = p[lane + 32 * t];
    s += v * v;
  }
#pragma unroll
  for (int off = 16; off; off >>= 1) s += __shfl_xor(s, off, 32);
  if (lane == 0) dst[row] = s;
}

// --------------------------------------------------- fp32 -> f16 pre-convert
__global__ void __launch_bounds__(256) cvt_f16_kernel(
    const float* __restrict__ X, const float* __restrict__ Y,
    _Float16* __restrict__ Xh, _Float16* __restrict__ Yh) {
  const float* src = blockIdx.y ? Y : X;
  _Float16* dst    = blockIdx.y ? Yh : Xh;
  size_t i = ((size_t)blockIdx.x * 256 + threadIdx.x) * 4;
  float4 f = *(const float4*)(src + i);
  v4h h;
  h[0] = (_Float16)f.x; h[1] = (_Float16)f.y;
  h[2] = (_Float16)f.z; h[3] = (_Float16)f.w;
  *(v4h*)(dst + i) = h;
}

// ------------------------------------------------------- fragment load (f16)
__device__ __forceinline__ v16h ldfrag(const _Float16* __restrict__ p0,
                                       const _Float16* __restrict__ p1) {
  v8h lo = *(const v8h*)p0;   // 16B aligned
  v8h hi = *(const v8h*)p1;
  return __builtin_shufflevector(lo, hi, 0, 1, 2, 3, 4, 5, 6, 7,
                                 8, 9, 10, 11, 12, 13, 14, 15);
}

// ---------------------------------------------- fp32 -> f16 inline (fallback)
__device__ __forceinline__ void cvt8(const float* __restrict__ p, v16h& d, int base) {
  float4 f0 = ((const float4*)p)[0];
  float4 f1 = ((const float4*)p)[1];
  d[base + 0] = (_Float16)f0.x; d[base + 1] = (_Float16)f0.y;
  d[base + 2] = (_Float16)f0.z; d[base + 3] = (_Float16)f0.w;
  d[base + 4] = (_Float16)f1.x; d[base + 5] = (_Float16)f1.y;
  d[base + 6] = (_Float16)f1.z; d[base + 7] = (_Float16)f1.w;
}

// ------------------------------------------------------------ epilogue helper
__device__ __forceinline__ void tile_epilogue(
    const v8f& c, int row0, int col0, int i, int j, int m, int kb,
    const float* __restrict__ x2, const float* __restrict__ y2,
    float* __restrict__ K) {
  int gcol = col0 + 16 * j + m;
  float yv = y2[gcol];
  int rbase = row0 + 16 * i + 8 * kb;
  float4 xa = ((const float4*)(x2 + rbase))[0];
  float4 xb = ((const float4*)(x2 + rbase))[1];
  float xs[8] = {xa.x, xa.y, xa.z, xa.w, xb.x, xb.y, xb.z, xb.w};
#pragma unroll
  for (int r = 0; r < 8; ++r) {
    // K = exp(-(x2 + y2 - 2*dot)/0.1) = exp((2*dot - x2 - y2) * 10)
    float val = __expf((2.0f * c[r] - xs[r] - yv) * 10.0f);
    K[(size_t)(rbase + r) * NN + gcol] = val;
  }
}

// ---------------- fast path: f16 panels, 32x64 tile per wave (2x4 WMMA frags)
__global__ void __launch_bounds__(256) kexp_f16_kernel(
    const _Float16* __restrict__ Xh, const _Float16* __restrict__ Yh,
    const float* __restrict__ x2, const float* __restrict__ y2,
    float* __restrict__ K) {
  int tid  = threadIdx.x;
  int lane = tid & 31;
  int w    = tid >> 5;          // 0..7
  int wr   = w >> 2;            // 0..1
  int wc   = w & 3;             // 0..3
  int row0 = (blockIdx.y * 2 + wr) * 32;
  int col0 = (blockIdx.x * 4 + wc) * 64;
  int m  = lane & 15;
  int kb = lane >> 4;           // 0 or 1

  v8f c0[4] = {{}, {}, {}, {}};
  v8f c1[4] = {{}, {}, {}, {}};

  const _Float16* A0 = Xh + (size_t)(row0 + m)      * DD;
  const _Float16* A1 = Xh + (size_t)(row0 + 16 + m) * DD;
  const _Float16* B0 = Yh + (size_t)(col0 + m)      * DD;
  const _Float16* B1 = Yh + (size_t)(col0 + 16 + m) * DD;
  const _Float16* B2 = Yh + (size_t)(col0 + 32 + m) * DD;
  const _Float16* B3 = Yh + (size_t)(col0 + 48 + m) * DD;
  const _Float16* B[4] = {B0, B1, B2, B3};

#pragma unroll
  for (int ks = 0; ks < DD; ks += 32) {
    // A frag (16x32 f16): lane m(+16kb): halfs0-7 = K[8kb..], halfs8-15 = K[16+8kb..]
    v16h a0 = ldfrag(A0 + ks + 8 * kb, A0 + ks + 16 + 8 * kb);
    v16h a1 = ldfrag(A1 + ks + 8 * kb, A1 + ks + 16 + 8 * kb);
#pragma unroll
    for (int j = 0; j < 4; ++j) {
      // B frag (32x16 f16): lane n(+16kb): halfs0-7 = K[8kb..], halfs8-15 = K[16+8kb..]
      v16h b = ldfrag(B[j] + ks + 8 * kb, B[j] + ks + 16 + 8 * kb);
      c0[j] = __builtin_amdgcn_wmma_f32_16x16x32_f16(false, a0, false, b, (short)0, c0[j], false, false);
      c1[j] = __builtin_amdgcn_wmma_f32_16x16x32_f16(false, a1, false, b, (short)0, c1[j], false, false);
    }
  }

#pragma unroll
  for (int j = 0; j < 4; ++j) {
    tile_epilogue(c0[j], row0, col0, 0, j, m, kb, x2, y2, K);
    tile_epilogue(c1[j], row0, col0, 1, j, m, kb, x2, y2, K);
  }
}

// ------------- fallback: inline-convert fp32, 32x32 tile per wave (2x2 frags)
__global__ void __launch_bounds__(256) kexp_kernel(
    const float* __restrict__ X, const float* __restrict__ Yt,
    const float* __restrict__ x2, const float* __restrict__ y2,
    float* __restrict__ K) {
  int tid  = threadIdx.x;
  int lane = tid & 31;
  int w    = tid >> 5;
  int wr   = w >> 2;
  int wc   = w & 3;
  int row0 = (blockIdx.y * 2 + wr) * 32;
  int col0 = (blockIdx.x * 4 + wc) * 32;
  int m  = lane & 15;
  int kb = lane >> 4;

  v8f c00 = {}, c01 = {}, c10 = {}, c11 = {};

  const float* A0 = X  + (size_t)(row0 + m)      * DD;
  const float* A1 = X  + (size_t)(row0 + 16 + m) * DD;
  const float* B0 = Yt + (size_t)(col0 + m)      * DD;
  const float* B1 = Yt + (size_t)(col0 + 16 + m) * DD;

#pragma unroll
  for (int ks = 0; ks < DD; ks += 32) {
    v16h a0, a1, b0, b1;
    cvt8(A0 + ks + 8 * kb,      a0, 0);
    cvt8(A0 + ks + 16 + 8 * kb, a0, 8);
    cvt8(A1 + ks + 8 * kb,      a1, 0);
    cvt8(A1 + ks + 16 + 8 * kb, a1, 8);
    cvt8(B0 + ks + 8 * kb,      b0, 0);
    cvt8(B0 + ks + 16 + 8 * kb, b0, 8);
    cvt8(B1 + ks + 8 * kb,      b1, 0);
    cvt8(B1 + ks + 16 + 8 * kb, b1, 8);

    c00 = __builtin_amdgcn_wmma_f32_16x16x32_f16(false, a0, false, b0, (short)0, c00, false, false);
    c01 = __builtin_amdgcn_wmma_f32_16x16x32_f16(false, a0, false, b1, (short)0, c01, false, false);
    c10 = __builtin_amdgcn_wmma_f32_16x16x32_f16(false, a1, false, b0, (short)0, c10, false, false);
    c11 = __builtin_amdgcn_wmma_f32_16x16x32_f16(false, a1, false, b1, (short)0, c11, false, false);
  }

  tile_epilogue(c00, row0, col0, 0, 0, m, kb, x2, y2, K);
  tile_epilogue(c01, row0, col0, 0, 1, m, kb, x2, y2, K);
  tile_epilogue(c10, row0, col0, 1, 0, m, kb, x2, y2, K);
  tile_epilogue(c11, row0, col0, 1, 1, m, kb, x2, y2, K);
}

// ------------------------------------------------------- rowsum -> u directly
__global__ void __launch_bounds__(256) rowsum_u_kernel(
    const float* __restrict__ K, float* __restrict__ u) {
  int wave = threadIdx.x >> 5;
  int lane = threadIdx.x & 31;
  int row  = blockIdx.x * 8 + wave;
  const float4* p = (const float4*)(K + (size_t)row * NN);
  float s = 0.f;
#pragma unroll 4
  for (int t = lane; t < NN / 4; t += 32) {
    float4 q = p[t];
    s += (q.x + q.y) + (q.z + q.w);
  }
#pragma unroll
  for (int off = 16; off; off >>= 1) s += __shfl_xor(s, off, 32);
  if (lane == 0) {
    u[row] = (1.0f / NN) / (s * (1.0f / NN) + 1e-8f);
  }
}

// ---------------------------------------- partial weighted colsums (Kt @ u)
__global__ void __launch_bounds__(256) colsum_partial_kernel(
    const float* __restrict__ K, const float* __restrict__ u,
    float* __restrict__ vacc) {
  int j  = blockIdx.x * 256 + threadIdx.x;   // column
  int i0 = blockIdx.y * 256;                 // row chunk
  const float* p = K + (size_t)i0 * NN + j;
  float acc = 0.f;
#pragma unroll 4
  for (int i = 0; i < 256; ++i) acc += p[(size_t)i * NN] * u[i0 + i];
  atomicAdd(&vacc[j], acc);
}

__global__ void __launch_bounds__(256) v_finalize_kernel(
    const float* __restrict__ vacc, float* __restrict__ v) {
  int j = blockIdx.x * 256 + threadIdx.x;
  v[j] = (1.0f / NN) / (vacc[j] + 1e-8f);
}

// ------------------------------------------- out = u[:,None] * K * v[None,:]
__global__ void __launch_bounds__(256) scale_kernel(
    float* __restrict__ K, const float* __restrict__ u,
    const float* __restrict__ v) {
  size_t idx = (size_t)blockIdx.x * 256 + threadIdx.x;  // one float4 / thread
  int row  = (int)(idx >> 11);        // 2048 float4 per row
  int c4   = (int)(idx & 2047);
  float us = u[row];
  float4 v4 = ((const float4*)v)[c4];
  float4 k4 = ((float4*)K)[idx];
  k4.x *= us * v4.x;
  k4.y *= us * v4.y;
  k4.z *= us * v4.z;
  k4.w *= us * v4.w;
  ((float4*)K)[idx] = k4;
}

// ---------------------------------------------------------------------------
extern "C" void kernel_launch(void* const* d_in, const int* in_sizes, int n_in,
                              void* d_out, int out_size, void* d_ws, size_t ws_size,
                              hipStream_t stream) {
  const float* src = (const float*)d_in[0];   // [8192, 256] fp32
  const float* tgt = (const float*)d_in[1];   // [8192, 256] fp32
  float* K = (float*)d_out;                   // [8192, 8192] fp32 (scratch + output)

  const size_t halfs_bytes = 2ull * NN * DD * sizeof(_Float16);  // 8 MB
  const size_t vec_bytes   = 5ull * NN * sizeof(float);          // 160 KB
  const bool fast = (ws_size >= halfs_bytes + vec_bytes);        // ws_size is fixed -> deterministic

  _Float16* Xh = (_Float16*)d_ws;
  _Float16* Yh = Xh + (size_t)NN * DD;
  float* vecs  = fast ? (float*)((char*)d_ws + halfs_bytes) : (float*)d_ws;
  float* x2   = vecs;
  float* y2   = vecs + NN;
  float* u    = vecs + 2 * NN;
  float* v    = vecs + 3 * NN;
  float* vacc = vecs + 4 * NN;

  hipMemsetAsync(vacc, 0, NN * sizeof(float), stream);
  rownorm2_kernel<<<dim3(NN / 8, 2), 256, 0, stream>>>(src, tgt, x2, y2);

  if (fast) {
    cvt_f16_kernel<<<dim3((NN * DD / 4) / 256, 2), 256, 0, stream>>>(src, tgt, Xh, Yh);
    // tiles: 32 rows x 64 cols per wave; block = 8 waves = 64 rows x 256 cols
    kexp_f16_kernel<<<dim3(NN / 256, NN / 64), 256, 0, stream>>>(Xh, Yh, x2, y2, K);
  } else {
    kexp_kernel<<<dim3(NN / 128, NN / 64), 256, 0, stream>>>(src, tgt, x2, y2, K);
  }

  rowsum_u_kernel<<<NN / 8, 256, 0, stream>>>(K, u);
  colsum_partial_kernel<<<dim3(NN / 256, NN / 256), 256, 0, stream>>>(K, u, vacc);
  v_finalize_kernel<<<NN / 256, 256, 0, stream>>>(vacc, v);
  scale_kernel<<<(NN / 4) * (NN / 256), 256, 0, stream>>>(K, u, v);
}
